// SpatialMTP1Hop_15436112462276
// MI455X (gfx1250) — compile-verified, matrix operand order checked by reading, not run on hardware
//
#include <hip/hip_runtime.h>
#include <math.h>

// ---- problem constants (from reference) ----
#define NN     100000
#define DEG    32
#define BCENT  8192
#define DMODEL 256
#define NHEAD  4
#define HDIM   64
#define DOUTV  128
#define LQ     33          // 1 + DEG rows per star
#define EPSF   1e-6f

// ---- WMMA vector types (gfx1250, wave32) ----
typedef __attribute__((ext_vector_type(16))) __bf16        v16bf;
typedef __attribute__((ext_vector_type(8)))  float         v8f;
typedef __attribute__((ext_vector_type(8)))  unsigned int  v8u;

// padded LDS row strides
#define XS  260            // Xres stride (floats)
#define AS  264            // Abuf/Act stride (ushorts)
#define QS  776            // QKV stride (ushorts)

// fragment-ordered bf16 weight blobs (written every launch; deterministic)
// qkv:128*768  o/g/u/d:128*256 each  head:128*128  => 245760 u32 (~1MB)
#define OFS_QKV  0
#define OFS_O    (OFS_QKV + 128*768)
#define OFS_G    (OFS_O   + 128*256)
#define OFS_U    (OFS_G   + 128*256)
#define OFS_D    (OFS_U   + 128*256)
#define OFS_H    (OFS_D   + 128*256)
#define BLOB_TOT (OFS_H   + 128*128)
__device__ unsigned int g_blob[BLOB_TOT];

__device__ __forceinline__ unsigned short f2bf(float x) {
    unsigned u = __float_as_uint(x);
    u += 0x7FFFu + ((u >> 16) & 1u);          // round-to-nearest-even
    return (unsigned short)(u >> 16);
}
__device__ __forceinline__ float bf2f(unsigned short h) {
    return __uint_as_float(((unsigned)h) << 16);
}
__device__ __forceinline__ float wred32(float v) {
#pragma unroll
    for (int o = 16; o > 0; o >>= 1) v += __shfl_xor(v, o, 32);
    return v;
}

// One output COLUMN of 16 cols (tile nt), all MT row-tiles, NB weight blobs,
// full K=256 reduction. Key property: each B fragment is loaded ONCE and
// reused for all MT row-tiles (3x less B traffic than job-per-(mt,nt));
// each A fragment is loaded once and feeds all NB blobs (gate+up fusion).
// A: LDS bf16 rows (row m feeds only D row m => no zero padding needed).
// B: fragment-ordered global blob [kt][nt][lane][8 u32] (2 coalesced b128/lane).
template<int MT, int NB>
__device__ __forceinline__ void wmma_col256(const unsigned short* __restrict__ A, int lda,
                                            const unsigned int* const (&blobs)[NB],
                                            int ntiles, int nt, int lane,
                                            v8f (&acc)[NB][MT])
{
    const int lo = lane & 15, hi = lane >> 4;
#pragma unroll
    for (int nb = 0; nb < NB; ++nb)
#pragma unroll
        for (int mt = 0; mt < MT; ++mt)
#pragma unroll
            for (int i = 0; i < 8; ++i) acc[nb][mt][i] = 0.f;

#pragma unroll
    for (int kt = 0; kt < 8; ++kt) {
        v8u bu[NB];
#pragma unroll
        for (int nb = 0; nb < NB; ++nb) {
            const unsigned int* bt = blobs[nb] + (((kt * ntiles + nt) * 32 + lane) << 3);
#pragma unroll
            for (int v = 0; v < 8; ++v) bu[nb][v] = bt[v];
        }
#pragma unroll
        for (int mt = 0; mt < MT; ++mt) {
            const unsigned int* arow =
                (const unsigned int*)(A + (mt * 16 + lo) * lda + kt * 32);
            v8u au;
#pragma unroll
            for (int v = 0; v < 4; ++v) au[v]     = arow[4 * hi + v];     // K = 8*hi+0..7
#pragma unroll
            for (int v = 0; v < 4; ++v) au[4 + v] = arow[8 + 4 * hi + v]; // K = 16+8*hi+0..7
            v16bf a = __builtin_bit_cast(v16bf, au);
#pragma unroll
            for (int nb = 0; nb < NB; ++nb) {
                v16bf b = __builtin_bit_cast(v16bf, bu[nb]);
                acc[nb][mt] = __builtin_amdgcn_wmma_f32_16x16x32_bf16(
                    false, a, false, b, (short)0, acc[nb][mt], false, false);
            }
        }
    }
}

// Pack W[O][256] f32 -> fragment-ordered bf16 pairs at g_blob[ofs + tid].
__global__ void pack_weight_kernel(const float* __restrict__ W, int O, unsigned ofs)
{
    int tid   = blockIdx.x * blockDim.x + threadIdx.x;
    int total = 128 * O;                       // (K/2)*O packed u32
    if (tid >= total) return;
    int ntiles = O >> 4;
    int v = tid & 7, lane = (tid >> 3) & 31, t = tid >> 8;
    int nt = t % ntiles, kt = t / ntiles;
    int lo = lane & 15, hi = lane >> 4;
    int n  = nt * 16 + lo;
    int kl = ((v < 4) ? 2 * v : 16 + 2 * (v - 4)) + 8 * hi;
    int k  = kt * 32 + kl;
    unsigned short a = f2bf(W[n * DMODEL + k]);
    unsigned short b = f2bf(W[n * DMODEL + k + 1]);
    g_blob[ofs + tid] = (unsigned)a | ((unsigned)b << 16);
}

__global__ void zero_out_kernel(float* out) { out[0] = 0.f; }

__global__ __launch_bounds__(256) void star_mtp_kernel(
    const float* __restrict__ H, const int* __restrict__ dst,
    const int* __restrict__ centers, const float* __restrict__ target,
    const float* __restrict__ ln_in_w, const float* __restrict__ ln1_w,
    const float* __restrict__ ln2_w,
    const float* __restrict__ b_qkv, const float* __restrict__ b_o,
    const float* __restrict__ b_head,
    float* __restrict__ out)
{
    __shared__ __align__(16) float          Xres[LQ * XS];      // residual stream (f32)
    __shared__ __align__(16) unsigned short Abuf[48 * AS];      // bf16 A operand (xn/ao/xn2/Z)
    __shared__ __align__(16) unsigned short QKV [LQ * QS];      // bf16 q|k|v ; reused as Act
    __shared__ int   idxS[LQ];
    __shared__ float redS;

    const int tid  = threadIdx.x;
    const int lane = tid & 31;
    const int lo   = lane & 15, hi = lane >> 4;
    const int wid  = tid >> 5;
    const int b    = blockIdx.x;
    const int center = centers[b];

    if (tid == 0) redS = 0.f;
    if (tid < LQ) idxS[tid] = (tid == 0) ? center : dst[center * DEG + (tid - 1)];
    __syncthreads();

    // ---- gather + rmsnorm(ln_in) -> Xres ; rmsnorm(ln1) -> Abuf (bf16) ----
    for (int r = wid; r < LQ; r += 8) {
        const float* hrow = H + (long)idxS[r] * DMODEL;
        float x[8], ss = 0.f;
#pragma unroll
        for (int i = 0; i < 8; ++i) { x[i] = hrow[lane + 32 * i]; ss += x[i] * x[i]; }
        ss = wred32(ss);
        float rms = rsqrtf(ss * (1.f / DMODEL) + EPSF);
        float y[8], ss2 = 0.f;
#pragma unroll
        for (int i = 0; i < 8; ++i) {
            int k = lane + 32 * i;
            y[i] = x[i] * rms * ln_in_w[k];
            Xres[r * XS + k] = y[i];
            ss2 += y[i] * y[i];
        }
        ss2 = wred32(ss2);
        float rms2 = rsqrtf(ss2 * (1.f / DMODEL) + EPSF);
#pragma unroll
        for (int i = 0; i < 8; ++i) {
            int k = lane + 32 * i;
            Abuf[r * AS + k] = f2bf(y[i] * rms2 * ln1_w[k]);
        }
    }
    __syncthreads();

    // ---- GEMM1: qkv = xn @ Wqkv^T + b_qkv -> QKV bf16 (rows < 33) ----
    {
        const unsigned int* const bl[1] = { g_blob + OFS_QKV };
        for (int nt = wid; nt < 48; nt += 8) {              // 768/16 col tiles
            v8f acc[1][3];
            wmma_col256<3, 1>(Abuf, AS, bl, 48, nt, lane, acc);
            int col = nt * 16 + lo;
            float bias = b_qkv[col];
#pragma unroll
            for (int mt = 0; mt < 3; ++mt)
#pragma unroll
                for (int r = 0; r < 8; ++r) {
                    int row = mt * 16 + r + 8 * hi;
                    if (row < LQ) QKV[row * QS + col] = f2bf(acc[0][mt][r] + bias);
                }
        }
    }
    __syncthreads();

    // ---- attention (~3% of FLOPs): one thread per (row, head), f32 ----
    if (tid < LQ * NHEAD) {
        int row = tid % LQ, h = tid / LQ;
        int qo = h * HDIM, ko = DMODEL + h * HDIM, vo = 2 * DMODEL + h * HDIM;
        float sc[LQ], mx = -1e30f;
        for (int j = 0; j < LQ; ++j) {
            float s = 0.f;
#pragma unroll 8
            for (int c0 = 0; c0 < HDIM; ++c0)
                s += bf2f(QKV[row * QS + qo + c0]) * bf2f(QKV[j * QS + ko + c0]);
            s *= 0.125f;                                    // 1/sqrt(64)
            sc[j] = s; mx = fmaxf(mx, s);
        }
        float den = 0.f;
        for (int j = 0; j < LQ; ++j) { sc[j] = __expf(sc[j] - mx); den += sc[j]; }
        float inv = 1.f / den;
        for (int c0 = 0; c0 < HDIM; c0 += 16) {
            float acc[16];
#pragma unroll
            for (int i = 0; i < 16; ++i) acc[i] = 0.f;
            for (int j = 0; j < LQ; ++j) {
                float p = sc[j];
#pragma unroll
                for (int i = 0; i < 16; ++i) acc[i] += p * bf2f(QKV[j * QS + vo + c0 + i]);
            }
#pragma unroll
            for (int i = 0; i < 16; ++i)
                Abuf[row * AS + qo + c0 + i] = f2bf(acc[i] * inv);  // ao overwrites xn
        }
    }
    __syncthreads();

    // ---- GEMM2: X += ao @ Wo^T + b_o ----
    {
        const unsigned int* const bl[1] = { g_blob + OFS_O };
        for (int nt = wid; nt < 16; nt += 8) {
            v8f acc[1][3];
            wmma_col256<3, 1>(Abuf, AS, bl, 16, nt, lane, acc);
            int col = nt * 16 + lo;
            float bias = b_o[col];
#pragma unroll
            for (int mt = 0; mt < 3; ++mt)
#pragma unroll
                for (int r = 0; r < 8; ++r) {
                    int row = mt * 16 + r + 8 * hi;
                    if (row < LQ) Xres[row * XS + col] += acc[0][mt][r] + bias;
                }
        }
    }
    __syncthreads();

    // ---- xn2 = rmsnorm(X, ln2) -> Abuf ----
    for (int r = wid; r < LQ; r += 8) {
        float y[8], ss = 0.f;
#pragma unroll
        for (int i = 0; i < 8; ++i) { y[i] = Xres[r * XS + lane + 32 * i]; ss += y[i] * y[i]; }
        ss = wred32(ss);
        float rms = rsqrtf(ss * (1.f / DMODEL) + EPSF);
#pragma unroll
        for (int i = 0; i < 8; ++i) {
            int k = lane + 32 * i;
            Abuf[r * AS + k] = f2bf(y[i] * rms * ln2_w[k]);
        }
    }
    __syncthreads();

    // ---- GEMM3 (fused): gate=xn2@Wg^T, up=xn2@Wu^T, act=silu(gate)*up -> Act ----
    unsigned short* Act = QKV;                              // reuse q|k|v region
    {
        const unsigned int* const bl[2] = { g_blob + OFS_G, g_blob + OFS_U };
        for (int nt = wid; nt < 16; nt += 8) {
            v8f acc[2][3];
            wmma_col256<3, 2>(Abuf, AS, bl, 16, nt, lane, acc);
            int col = nt * 16 + lo;
#pragma unroll
            for (int mt = 0; mt < 3; ++mt)
#pragma unroll
                for (int r = 0; r < 8; ++r) {
                    int row = mt * 16 + r + 8 * hi;
                    if (row < LQ) {
                        float g = acc[0][mt][r];
                        Act[row * AS + col] =
                            f2bf(g * (1.f / (1.f + __expf(-g))) * acc[1][mt][r]);
                    }
                }
        }
    }
    __syncthreads();

    // ---- GEMM4: X += act @ Wd^T  (Z = X afterwards) ----
    {
        const unsigned int* const bl[1] = { g_blob + OFS_D };
        for (int nt = wid; nt < 16; nt += 8) {
            v8f acc[1][3];
            wmma_col256<3, 1>(Act, AS, bl, 16, nt, lane, acc);
            int col = nt * 16 + lo;
#pragma unroll
            for (int mt = 0; mt < 3; ++mt)
#pragma unroll
                for (int r = 0; r < 8; ++r) {
                    int row = mt * 16 + r + 8 * hi;
                    if (row < LQ) Xres[row * XS + col] += acc[0][mt][r];
                }
        }
    }
    __syncthreads();

    // ---- head A-operand: rows 0..31 = bf16(Z[1..32]) ----
    for (int i = tid; i < 32 * DMODEL; i += 256) {
        int row = i >> 8, col = i & 255;
        Abuf[row * AS + col] = f2bf(Xres[(row + 1) * XS + col]);
    }
    __syncthreads();

    // ---- GEMM5: y_hat = Z[1:] @ Wh^T + b_head ; fused MSE loss ----
    float lacc = 0.f;
    {
        const unsigned int* const bl[1] = { g_blob + OFS_H };
        for (int nt = wid; nt < 8; nt += 8) {               // 128/16: one col per wave
            v8f acc[1][2];
            wmma_col256<2, 1>(Abuf, AS, bl, 8, nt, lane, acc);
            int col = nt * 16 + lo;
            float bias = b_head[col];
#pragma unroll
            for (int mt = 0; mt < 2; ++mt)
#pragma unroll
                for (int r = 0; r < 8; ++r) {
                    int row  = mt * 16 + r + 8 * hi;        // 0..31 (neighbor index)
                    int node = idxS[row + 1];
                    float d  = acc[0][mt][r] + bias - target[(long)node * DOUTV + col];
                    lacc += d * d;
                }
        }
    }
    lacc = wred32(lacc);
    if (lane == 0) atomicAdd(&redS, lacc);
    __syncthreads();
    if (tid == 0) {
        // mean over DOUT (/128), /deg (=32), mean over B (/8192)
        const float scale = 1.f / ((float)DOUTV * (float)DEG * (float)BCENT);
        atomicAdd(out, redS * scale);
    }
}

extern "C" void kernel_launch(void* const* d_in, const int* in_sizes, int n_in,
                              void* d_out, int out_size, void* d_ws, size_t ws_size,
                              hipStream_t stream)
{
    (void)in_sizes; (void)n_in; (void)out_size; (void)d_ws; (void)ws_size;
    const float* H       = (const float*)d_in[0];
    const int*   ei      = (const int*)d_in[1];
    const int*   centers = (const int*)d_in[2];
    const float* target  = (const float*)d_in[3];
    /* d_in[4] = max_deg (==DEG, every node has exactly DEG edges) */
    const float* ln_in_w = (const float*)d_in[5];
    const float* ln1_w   = (const float*)d_in[6];
    const float* ln2_w   = (const float*)d_in[7];
    const float* w_qkv   = (const float*)d_in[8];
    const float* b_qkv   = (const float*)d_in[9];
    const float* w_o     = (const float*)d_in[10];
    const float* b_o     = (const float*)d_in[11];
    const float* w_gate  = (const float*)d_in[12];
    const float* w_up    = (const float*)d_in[13];
    const float* w_down  = (const float*)d_in[14];
    const float* w_head  = (const float*)d_in[15];
    const float* b_head  = (const float*)d_in[16];

    const int* dst = ei + (size_t)NN * DEG;   // edge_index row 1

    zero_out_kernel<<<1, 1, 0, stream>>>((float*)d_out);

    pack_weight_kernel<<<(128*768 + 255) / 256, 256, 0, stream>>>(w_qkv,  768, OFS_QKV);
    pack_weight_kernel<<<(128*256 + 255) / 256, 256, 0, stream>>>(w_o,    256, OFS_O);
    pack_weight_kernel<<<(128*256 + 255) / 256, 256, 0, stream>>>(w_gate, 256, OFS_G);
    pack_weight_kernel<<<(128*256 + 255) / 256, 256, 0, stream>>>(w_up,   256, OFS_U);
    pack_weight_kernel<<<(128*256 + 255) / 256, 256, 0, stream>>>(w_down, 256, OFS_D);
    pack_weight_kernel<<<(128*128 + 255) / 256, 256, 0, stream>>>(w_head, 128, OFS_H);

    star_mtp_kernel<<<BCENT, 256, 0, stream>>>(H, dst, centers, target,
                                               ln_in_w, ln1_w, ln2_w,
                                               b_qkv, b_o, b_head,
                                               (float*)d_out);
}
